// Qwen3Attention_83691732729874
// MI455X (gfx1250) — matrix operand order repaired and checked by reading, not compiled, not run on hardware
//
#include <hip/hip_runtime.h>

// ---------------------------------------------------------------------------
// Qwen3 attention block for MI455X (gfx1250, wave32, WMMA bf16 16x16x32).
// All GEMMs (QKV proj, QK^T, PV, out proj) run on v_wmma_f32_16x16x32_bf16.
// Attention stages K/V tiles cooperatively into LDS with the gfx1250
// async-to-LDS path (GLOBAL_LOAD_ASYNC_TO_LDS_B128 + s_wait_asynccnt).
// ---------------------------------------------------------------------------

typedef __attribute__((ext_vector_type(16))) __bf16 v16bf;
typedef __attribute__((ext_vector_type(8)))  __bf16 v8bf;
typedef __attribute__((ext_vector_type(8)))  float  v8f;
typedef __attribute__((__vector_size__(16))) int    v4i;

#define HIDDEN 1024
#define NH     16
#define NKV    8
#define HD     128
#define SEQ    2048
#define BATCH  2
#define NTOK   (BATCH * SEQ)   // 4096

union UFrag { v16bf v; v8bf h2[2]; };

__device__ __forceinline__ v8f wmma_bf16(v16bf a, v16bf b, v8f c) {
  // (neg_a, A, neg_b, B, c_mod, C, reuse_a, reuse_b)
  return __builtin_amdgcn_wmma_f32_16x16x32_bf16(false, a, false, b,
                                                 (short)0, c, false, false);
}

// --- async global->LDS 16B copy with always-compiling fallback --------------
#define AS1 __attribute__((address_space(1)))
#define AS3 __attribute__((address_space(3)))

__device__ __forceinline__ void copy16_to_lds(const void* g, void* l) {
#if __has_builtin(__builtin_amdgcn_global_load_async_to_lds_b128)
  __builtin_amdgcn_global_load_async_to_lds_b128(
      (AS1 v4i*)g, (AS3 v4i*)l, 0, 0);
#else
  *(int4*)l = *(const int4*)g;   // synchronous fallback
#endif
}

__device__ __forceinline__ void copy_to_lds_fence() {
#if __has_builtin(__builtin_amdgcn_global_load_async_to_lds_b128)
#if __has_builtin(__builtin_amdgcn_s_wait_asynccnt)
  __builtin_amdgcn_s_wait_asynccnt(0);
#else
  asm volatile("s_wait_asynccnt 0x0" ::: "memory");
#endif
#endif
}

// ---------------------------------------------------------------------------
// Generic C(f32) = A(f32 as bf16) @ B(f32 as bf16) GEMM.
// Block tile 128x128, 8 waves, each wave owns a 16x128 slab (8 accumulators).
// K stepped by 32 through LDS (A row-major [m][k], B transposed [n][k]).
// f32->bf16 conversion happens while staging, so staging stays VALU-based.
// ---------------------------------------------------------------------------
#define BM 128
#define BN 128
#define BK 32
#define LDSK 48   // padded row stride (bf16 elems): 96B keeps 16B/32B alignment

__global__ __launch_bounds__(256) void gemm_bf16_wmma(
    const float* __restrict__ A, const float* __restrict__ B,
    float* __restrict__ C, int M, int N, int K)
{
  __shared__ alignas(32) __bf16 As[BM][LDSK];
  __shared__ alignas(32) __bf16 Bs[BN][LDSK];

  const int tid  = threadIdx.x;
  const int lane = tid & 31;
  const int wave = tid >> 5;
  const int half = lane >> 4;
  const int l16  = lane & 15;
  const int n0   = blockIdx.x * BN;
  const int m0   = blockIdx.y * BM;
  (void)M; (void)lane;

  v8f acc[8] = {};

  for (int k0 = 0; k0 < K; k0 += BK) {
    // A tile: 128x32 f32 -> bf16 (float4 loads, 4 per thread)
    #pragma unroll
    for (int i = 0; i < 4; ++i) {
      int idx4 = tid + i * 256;          // 0..1023 float4's
      int row  = idx4 >> 3;
      int kk   = (idx4 & 7) << 2;
      const float4 v = *(const float4*)&A[(size_t)(m0 + row) * K + k0 + kk];
      As[row][kk + 0] = (__bf16)v.x;
      As[row][kk + 1] = (__bf16)v.y;
      As[row][kk + 2] = (__bf16)v.z;
      As[row][kk + 3] = (__bf16)v.w;
    }
    // B tile: 32x128 f32 -> bf16, stored transposed [n][k]
    #pragma unroll
    for (int i = 0; i < 16; ++i) {
      int idx = tid + i * 256;           // 0..4095
      int kk  = idx >> 7;
      int nn  = idx & 127;
      Bs[nn][kk] = (__bf16)B[(size_t)(k0 + kk) * N + n0 + nn];
    }
    __syncthreads();

    // A fragment: lane L -> row (wave*16 + l16), K = 16*half + i (contiguous)
    UFrag af;
    const int arow = wave * 16 + l16;
    af.h2[0] = *(const v8bf*)&As[arow][16 * half];
    af.h2[1] = *(const v8bf*)&As[arow][16 * half + 8];

    #pragma unroll
    for (int n = 0; n < 8; ++n) {
      // B fragment: lane L -> col (n*16 + l16), K = 16*half + i (contig 32B)
      const v16bf bf = *(const v16bf*)&Bs[n * 16 + l16][16 * half];
      acc[n] = wmma_bf16(af.v, bf, acc[n]);
    }
    __syncthreads();
  }

  // C/D layout: VGPR e, row = 8*half + e, col = l16
  #pragma unroll
  for (int n = 0; n < 8; ++n) {
    #pragma unroll
    for (int e = 0; e < 8; ++e) {
      int row = m0 + wave * 16 + half * 8 + e;
      int col = n0 + n * 16 + l16;
      C[(size_t)row * N + col] = acc[n][e];
    }
  }
}

// ---------------------------------------------------------------------------
// Fused per-head RMSNorm + RoPE. One wave per (b, s, head); heads 0..15 = Q,
// 16..23 = K. 128 dims -> 4 per lane; RoPE partner d^64 is i^2 (in-lane).
// Emits bf16 in attention layout [b][h][s][d].
// ---------------------------------------------------------------------------
__global__ __launch_bounds__(256) void rmsnorm_rope_kernel(
    const float* __restrict__ qf, const float* __restrict__ kf,
    const int* __restrict__ positions,
    const float* __restrict__ qw, const float* __restrict__ kw,
    __bf16* __restrict__ Qh, __bf16* __restrict__ Kh)
{
  const int lane = threadIdx.x & 31;
  const int wave = threadIdx.x >> 5;
  const int gid  = blockIdx.x * 8 + wave;          // < B*S*24
  const int b    = gid / (SEQ * 24);
  const int rem  = gid % (SEQ * 24);
  const int s    = rem / 24;
  const int hh   = rem % 24;
  const bool isq = hh < 16;

  const float* src = isq ? &qf[((size_t)(b * SEQ + s) * NH + hh) * HD]
                         : &kf[((size_t)(b * SEQ + s) * NKV + (hh - 16)) * HD];
  const float* w = isq ? qw : kw;

  float x[4];
  float ss = 0.f;
  #pragma unroll
  for (int i = 0; i < 4; ++i) { x[i] = src[lane + 32 * i]; ss += x[i] * x[i]; }
  #pragma unroll
  for (int off = 16; off >= 1; off >>= 1) ss += __shfl_xor(ss, off);
  const float inv = rsqrtf(ss * (1.0f / HD) + 1e-6f);

  float xn[4];
  #pragma unroll
  for (int i = 0; i < 4; ++i) xn[i] = x[i] * inv * w[lane + 32 * i];

  const float pos = (float)positions[s];
  __bf16* dst = isq ? &Qh[((size_t)(b * NH + hh) * SEQ + s) * HD]
                    : &Kh[((size_t)(b * NKV + (hh - 16)) * SEQ + s) * HD];
  const float LOG_THETA = 13.815510557964274f;     // ln(1e6)
  #pragma unroll
  for (int i = 0; i < 4; ++i) {
    int d  = lane + 32 * i;
    int fd = d & 63;
    float invf = __expf(-(float)fd * (LOG_THETA / 64.0f));
    float ang  = pos * invf;
    float c = __cosf(ang), sn = __sinf(ang);
    float rot = (i < 2) ? -xn[i ^ 2] : xn[i ^ 2];  // rotate_half: partner d^64
    dst[d] = (__bf16)(xn[i] * c + rot * sn);
  }
}

// ---------------------------------------------------------------------------
// V: f32 [b][s][kv][d] -> bf16 transposed [b][kv][d][s] so PV B-fragments
// (16 consecutive keys per lane) are contiguous 32B loads.
// ---------------------------------------------------------------------------
__global__ __launch_bounds__(256) void v_transpose_kernel(
    const float* __restrict__ vf, __bf16* __restrict__ Vt)
{
  const int idx = blockIdx.x * 256 + threadIdx.x;  // < B*S*NKV*HD
  const int d  = idx & 127;
  const int t  = idx >> 7;
  const int kv = t & 7;
  const int t2 = t >> 3;
  const int s  = t2 & (SEQ - 1);
  const int b  = t2 >> 11;
  Vt[(((size_t)(b * NKV + kv) * HD) + d) * SEQ + s] = (__bf16)vf[idx];
}

// ---------------------------------------------------------------------------
// Flash attention, cooperative block version.
// One block = 4 waves = same (b, head), 4 consecutive 16-query tiles.
// Per 32-key step: all 128 threads stage the shared K tile (32x128) and
// V tile (128x32, d-major) into LDS via async-to-LDS, fence with
// s_wait_asynccnt, then each active wave runs 8 WMMAs for S = Q K^T,
// online softmax (half-wave xor reductions match the C-layout rows),
// P redistributed C->A layout via a per-wave LDS tile, and 8 WMMAs for
// O += P V. Waves whose causal range is exhausted still co-fill and hit
// both barriers (wave-uniform guard, EXEC all-ones around WMMA).
// ---------------------------------------------------------------------------
#define AW 4            // waves per block
#define KLD 144         // K tile row stride (bf16): 288B, 32B-aligned, padded
#define VLD 48          // V tile row stride (bf16): 96B, 32B-aligned, padded

__global__ __launch_bounds__(128) void flash_attn_kernel(
    const __bf16* __restrict__ Qh, const __bf16* __restrict__ Kh,
    const __bf16* __restrict__ Vt, float* __restrict__ ctx)
{
  __shared__ alignas(32) __bf16 Ktile[32][KLD];    // [key_local][d]
  __shared__ alignas(32) __bf16 Vtile[HD][VLD];    // [d][key_local]
  __shared__ alignas(32) __bf16 pbuf[AW][16][48];

  const int tid  = threadIdx.x;
  const int lane = tid & 31;
  const int wave = tid >> 5;
  const int half = lane >> 4;
  const int l16  = lane & 15;

  const int tile = blockIdx.x * AW + wave;         // < B*NH*(S/16) = 4096
  const int b    = tile >> 11;
  const int r    = tile & 2047;
  const int h    = r >> 7;
  const int qt   = r & 127;
  const int q0   = qt * 16;
  const int kvh  = h >> 1;                         // GQA: 2 q-heads per kv-head
  const int qblk0 = ((blockIdx.x * AW) & 127) * 16;  // first q0 in this block
  const int jend  = qblk0 + 16 * AW;               // keys needed by last wave

  const __bf16* Qp = &Qh[((size_t)(b * NH + h) * SEQ + q0) * HD];
  const __bf16* Kp = &Kh[(size_t)(b * NKV + kvh) * SEQ * HD];
  const __bf16* Vp = &Vt[(size_t)(b * NKV + kvh) * HD * SEQ];

  // Q fragments (A layout), head dim split into 4 chunks of 32
  v16bf qfrag[4];
  #pragma unroll
  for (int c = 0; c < 4; ++c) {
    UFrag u;
    const __bf16* row = &Qp[(size_t)l16 * HD + 32 * c];
    u.h2[0] = *(const v8bf*)&row[16 * half];
    u.h2[1] = *(const v8bf*)&row[16 * half + 8];
    qfrag[c] = u.v;
  }

  v8f acc[8] = {};
  float mrow[8], lrow[8];
  #pragma unroll
  for (int e = 0; e < 8; ++e) { mrow[e] = -1e30f; lrow[e] = 0.0f; }

  const float scale = 0.08838834764831843f;        // 1/sqrt(128)

  for (int j0 = 0; j0 < jend; j0 += 32) {          // keys j0 .. j0+31
    // ---- cooperative fill: K tile 32x128 (512 x 16B), V tile 128x32 ----
    #pragma unroll
    for (int i = 0; i < 4; ++i) {
      int idx = tid + i * 128;                     // 0..511
      int kr  = idx >> 4;                          // key row 0..31
      int kc  = idx & 15;                          // 16B chunk 0..15
      copy16_to_lds(&Kp[(size_t)(j0 + kr) * HD + kc * 8], &Ktile[kr][kc * 8]);
      int vr  = idx >> 2;                          // d row 0..127
      int vc  = idx & 3;                           // 16B chunk 0..3
      copy16_to_lds(&Vp[(size_t)vr * SEQ + j0 + vc * 8], &Vtile[vr][vc * 8]);
    }
    copy_to_lds_fence();
    __syncthreads();

    if (j0 < q0 + 16) {                            // wave-uniform causal guard
      v8f s0 = {}, s1 = {};
      #pragma unroll
      for (int c = 0; c < 4; ++c) {
        const v16bf bk0 = *(const v16bf*)&Ktile[l16][32 * c + 16 * half];
        s0 = wmma_bf16(qfrag[c], bk0, s0);
        const v16bf bk1 = *(const v16bf*)&Ktile[16 + l16][32 * c + 16 * half];
        s1 = wmma_bf16(qfrag[c], bk1, s1);
      }

      // scale + causal mask (VALU selects, EXEC stays all-ones for WMMA)
      const bool need_mask = (j0 + 32 > q0);
      #pragma unroll
      for (int e = 0; e < 8; ++e) {
        float a  = s0[e] * scale;
        float bb = s1[e] * scale;
        if (need_mask) {
          int rowg = q0 + half * 8 + e;
          if (j0 + l16 > rowg)      a  = -1e30f;
          if (j0 + 16 + l16 > rowg) bb = -1e30f;
        }
        s0[e] = a; s1[e] = bb;
      }

      // online softmax (row reductions across the 16-lane half: xor 1..8)
      float p0[8], p1[8], sc[8];
      #pragma unroll
      for (int e = 0; e < 8; ++e) {
        float mx = fmaxf(s0[e], s1[e]);
        #pragma unroll
        for (int off = 8; off >= 1; off >>= 1)
          mx = fmaxf(mx, __shfl_xor(mx, off));
        float mnew = fmaxf(mrow[e], mx);
        float f    = __expf(mrow[e] - mnew);
        p0[e] = __expf(s0[e] - mnew);
        p1[e] = __expf(s1[e] - mnew);
        float sum = p0[e] + p1[e];
        #pragma unroll
        for (int off = 8; off >= 1; off >>= 1) sum += __shfl_xor(sum, off);
        lrow[e] = lrow[e] * f + sum;
        mrow[e] = mnew;
        sc[e]   = f;
      }
      #pragma unroll
      for (int n = 0; n < 8; ++n)
        #pragma unroll
        for (int e = 0; e < 8; ++e) acc[n][e] *= sc[e];

      // P: C-layout -> bf16 A-layout through per-wave LDS tile (16x32)
      #pragma unroll
      for (int e = 0; e < 8; ++e) {
        pbuf[wave][half * 8 + e][l16]      = (__bf16)p0[e];
        pbuf[wave][half * 8 + e][16 + l16] = (__bf16)p1[e];
      }
      asm volatile("s_wait_dscnt 0x0" ::: "memory");
      UFrag pu;
      pu.h2[0] = *(const v8bf*)&pbuf[wave][l16][16 * half];
      pu.h2[1] = *(const v8bf*)&pbuf[wave][l16][16 * half + 8];
      const v16bf pfrag = pu.v;

      // O += P @ V (B fragment: col d = 16n+l16, keys 16*half+i contiguous)
      #pragma unroll
      for (int n = 0; n < 8; ++n) {
        const v16bf vfrag = *(const v16bf*)&Vtile[16 * n + l16][16 * half];
        acc[n] = wmma_bf16(pfrag, vfrag, acc[n]);
      }
    }
    __syncthreads();                               // protect tiles before refill
  }

  // normalize and store ctx f32 [b][s][h*128+d] (A matrix of out-proj GEMM)
  #pragma unroll
  for (int n = 0; n < 8; ++n) {
    #pragma unroll
    for (int e = 0; e < 8; ++e) {
      int rowg = q0 + half * 8 + e;
      int d    = 16 * n + l16;
      ctx[(size_t)(b * SEQ + rowg) * (NH * HD) + h * HD + d] =
          acc[n][e] / lrow[e];
    }
  }
}

// ---------------------------------------------------------------------------
// Host-side orchestration
// ---------------------------------------------------------------------------
extern "C" void kernel_launch(void* const* d_in, const int* in_sizes, int n_in,
                              void* d_out, int out_size, void* d_ws, size_t ws_size,
                              hipStream_t stream) {
  (void)in_sizes; (void)n_in; (void)out_size; (void)ws_size;

  const float* x   = (const float*)d_in[0];
  const int*   pos = (const int*)  d_in[1];
  // d_in[2] = attn_mask (causal; implemented analytically in-kernel)
  const float* wq  = (const float*)d_in[3];
  const float* wk  = (const float*)d_in[4];
  const float* wv  = (const float*)d_in[5];
  const float* wo  = (const float*)d_in[6];
  const float* qnw = (const float*)d_in[7];
  const float* knw = (const float*)d_in[8];

  char* ws = (char*)d_ws;
  const size_t MB = 1024ull * 1024ull;
  float*  qf  = (float*) (ws + 0);        // 4096x2048 f32  (32 MB)
  float*  kf  = (float*) (ws + 32 * MB);  // 4096x1024 f32  (16 MB)
  float*  vf  = (float*) (ws + 48 * MB);  // 4096x1024 f32  (16 MB)
  __bf16* Qh  = (__bf16*)(ws + 64 * MB);  // [2][16][2048][128] bf16 (16 MB)
  __bf16* Kh  = (__bf16*)(ws + 80 * MB);  // [2][8][2048][128]  bf16 ( 8 MB)
  __bf16* Vt  = (__bf16*)(ws + 88 * MB);  // [2][8][128][2048]  bf16 ( 8 MB)
  float*  ctx = qf;                       // alias: qf consumed before attention

  dim3 blk(256);
  // QKV projections: x(4096x1024) @ w -> f32
  gemm_bf16_wmma<<<dim3(2048 / BN, NTOK / BM), blk, 0, stream>>>(
      x, wq, qf, NTOK, NH * HD, HIDDEN);
  gemm_bf16_wmma<<<dim3(1024 / BN, NTOK / BM), blk, 0, stream>>>(
      x, wk, kf, NTOK, NKV * HD, HIDDEN);
  gemm_bf16_wmma<<<dim3(1024 / BN, NTOK / BM), blk, 0, stream>>>(
      x, wv, vf, NTOK, NKV * HD, HIDDEN);

  // per-head RMSNorm + RoPE -> bf16 attention layouts
  rmsnorm_rope_kernel<<<(NTOK * 24) / 8, 256, 0, stream>>>(
      qf, kf, pos, qnw, knw, Qh, Kh);
  v_transpose_kernel<<<(NTOK * NKV * HD) / 256, 256, 0, stream>>>(vf, Vt);

  // flash attention -> ctx f32 (4096x2048)
  flash_attn_kernel<<<(BATCH * NH * (SEQ / 16)) / AW, 32 * AW, 0, stream>>>(
      Qh, Kh, Vt, ctx);

  // output projection: ctx(4096x2048) @ wo(2048x1024) -> d_out f32
  gemm_bf16_wmma<<<dim3(1024 / BN, NTOK / BM), blk, 0, stream>>>(
      ctx, wo, (float*)d_out, NTOK, HIDDEN, NH * HD);
}
